// BilinearScoringFunction_86251533238870
// MI455X (gfx1250) — matrix operand order repaired and checked by reading, not compiled, not run on hardware
//
#include <hip/hip_runtime.h>
#include <hip/hip_bf16.h>

#define B_N 4096
#define D_N 512
#define H_N 256

typedef __attribute__((ext_vector_type(16))) __bf16 v16bf;
typedef __attribute__((ext_vector_type(8)))  __bf16 v8bf;
typedef __attribute__((ext_vector_type(8)))  float  v8f;

__device__ __forceinline__ unsigned short f32_to_bf16(float x) {
    unsigned int u = __float_as_uint(x);
    unsigned int r = u + 0x7FFFu + ((u >> 16) & 1u);  // round-to-nearest-even
    return (unsigned short)(r >> 16);
}

__global__ void zero_out_kernel(float* __restrict__ out, int n) {
    int i = blockIdx.x * blockDim.x + threadIdx.x;
    if (i < n) out[i] = 0.0f;
}

__global__ void convert_head_kernel(const float* __restrict__ src,
                                    unsigned short* __restrict__ dst, int n) {
    int i = blockIdx.x * blockDim.x + threadIdx.x;
    if (i < n) dst[i] = f32_to_bf16(src[i]);
}

// W[h][i][j] fp32  ->  Wt[h][j][i] bf16  (32x32 LDS tile transpose, coalesced R/W)
__global__ void transpose_convert_W_kernel(const float* __restrict__ W,
                                           unsigned short* __restrict__ Wt) {
    __shared__ unsigned short tile[32][33];
    const int h  = blockIdx.z;
    const int i0 = blockIdx.y * 32;
    const int j0 = blockIdx.x * 32;
    const int tx = threadIdx.x;   // 0..31
    const int ty = threadIdx.y;   // 0..7
    const size_t hbase = (size_t)h * D_N * D_N;
#pragma unroll
    for (int k = 0; k < 4; ++k) {
        int i = ty + 8 * k;
        tile[i][tx] = f32_to_bf16(W[hbase + (size_t)(i0 + i) * D_N + (j0 + tx)]);
    }
    __syncthreads();
#pragma unroll
    for (int k = 0; k < 4; ++k) {
        int j = ty + 8 * k;
        Wt[((size_t)h * D_N + (j0 + j)) * D_N + (i0 + tx)] = tile[tx][j];
    }
}

// Lin[b][h] = b_R[h] + [head|tail] . V_R[h]   (small side GEMV, ~1 GMAC total)
__global__ void linear_kernel(const float* __restrict__ head, const float* __restrict__ tail,
                              const float* __restrict__ V, const float* __restrict__ bR,
                              float* __restrict__ Lin) {
    const int h = blockIdx.y;
    const int b = blockIdx.x * blockDim.x + threadIdx.x;
    const float* vh = V + (size_t)h * (2 * D_N);
    const float* vt = vh + D_N;
    const float* hb = head + (size_t)b * D_N;
    const float* tb = tail + (size_t)b * D_N;
    float s = bR[h];
    for (int k = 0; k < D_N; ++k) s = fmaf(hb[k], vh[k], s);
    for (int k = 0; k < D_N; ++k) s = fmaf(tb[k], vt[k], s);
    Lin[(size_t)b * H_N + h] = s;
}

// One wave per (16-batch tile, h). tmp = head_tile x W[h] via bf16 WMMA (f32 acc),
// B-fragment stream software-pipelined through an 8-deep register ring so WMMAs
// issue back-to-back behind partial load waits instead of s_wait_loadcnt 0.
__global__ void __launch_bounds__(128) bilinear_wmma_kernel(
        const unsigned short* __restrict__ headBF,
        const unsigned short* __restrict__ WtBF,
        const float* __restrict__ tail,
        const float* __restrict__ Lin,
        const float* __restrict__ uR,
        float* __restrict__ out) {
    const int h    = blockIdx.x;
    const int wave = threadIdx.x >> 5;
    const int lane = threadIdx.x & 31;
    const int half = lane >> 4;     // which 16-lane half
    const int n    = lane & 15;     // column / row-in-tile index
    const int b0   = (blockIdx.y * 4 + wave) * 16;

    const __bf16* headb = (const __bf16*)headBF;
    const __bf16* Wt    = (const __bf16*)WtBF;

    // A fragments (contiguous k = 16*half + s, same order as B) — hoisted, reused by all 32 j-tiles.
    v16bf A[16];
    {
        const __bf16* ap = headb + (size_t)(b0 + n) * D_N + half * 16;
#pragma unroll
        for (int kt = 0; kt < 16; ++kt) {
            v8bf lo = *(const v8bf*)(ap + kt * 32);
            v8bf hi = *(const v8bf*)(ap + kt * 32 + 8);
            A[kt] = __builtin_shufflevector(lo, hi,
                    0,1,2,3,4,5,6,7,8,9,10,11,12,13,14,15);
        }
    }

    // B fragment: 16 contiguous bf16 per lane (col n, K-chunk) thanks to transpose.
    const __bf16* wbase = Wt + ((size_t)h * D_N + n) * D_N + half * 16;
    const float*  tbase = tail + (size_t)(b0 + half * 8) * D_N + n;

    // Prologue: fill 8-deep B ring (fragments t = 0..7) and first tail vector.
    v16bf Bbuf[8];
#pragma unroll
    for (int t = 0; t < 8; ++t)
        Bbuf[t] = *(const v16bf*)(wbase + t * 32);

    float tcur[8];
#pragma unroll
    for (int v = 0; v < 8; ++v) tcur[v] = tbase[(size_t)v * D_N];

    float s[8] = {0.f, 0.f, 0.f, 0.f, 0.f, 0.f, 0.f, 0.f};

#pragma unroll 1
    for (int jt = 0; jt < 32; ++jt) {
        const __bf16* wp  = wbase + (size_t)jt * (16 * D_N);
        const int     jn  = (jt < 31) ? (jt + 1) : 31;   // clamp: stay in-bounds
        const __bf16* wpn = wbase + (size_t)jn * (16 * D_N);
        // WGP-scope prefetch (locality 3) of the tile two j-steps ahead: pulls the
        // L2-resident W stream into WGP$/L0 so the ring's partial waits never stall.
        __builtin_prefetch(wbase + (size_t)((jt < 30) ? jt + 2 : 31) * (16 * D_N), 0, 3);

        // Preload next tile's tail scalars (double buffer).
        float tnext[8];
        const float* tpn = tbase + (size_t)jn * 16;
#pragma unroll
        for (int v = 0; v < 8; ++v) tnext[v] = tpn[(size_t)v * D_N];

        v8f acc = {0.f, 0.f, 0.f, 0.f, 0.f, 0.f, 0.f, 0.f};
        // Invariant: at t = jt*16+kt, Bbuf[t&7] holds fragment t; after the WMMA
        // consumes it we refill that slot with fragment t+8 (8 fragments / 16
        // b128 loads always in flight -> partial waits, WMMAs stay back-to-back).
#pragma unroll
        for (int kt = 0; kt < 16; ++kt) {
            acc = __builtin_amdgcn_wmma_f32_16x16x32_bf16(
                      false, A[kt], false, Bbuf[kt & 7], (short)0, acc, false, false);
            if (kt < 8)
                Bbuf[kt & 7] = *(const v16bf*)(wp  + (kt + 8) * 32);
            else
                Bbuf[kt & 7] = *(const v16bf*)(wpn + (kt - 8) * 32);
        }

        // C/D layout: VGPR v, lane -> (row = v + 8*half, col = n). Fold tail dot.
#pragma unroll
        for (int v = 0; v < 8; ++v) s[v] = fmaf(acc[v], tcur[v], s[v]);
#pragma unroll
        for (int v = 0; v < 8; ++v) tcur[v] = tnext[v];
    }

    // Reduce over the 16 columns held by each half-wave.
#pragma unroll
    for (int v = 0; v < 8; ++v) {
        float x = s[v];
        x += __shfl_xor(x, 1, 16);
        x += __shfl_xor(x, 2, 16);
        x += __shfl_xor(x, 4, 16);
        x += __shfl_xor(x, 8, 16);
        s[v] = x;
    }

    if (n == 0) {
        const float uh = uR[h];
#pragma unroll
        for (int v = 0; v < 8; ++v) {
            int b = b0 + half * 8 + v;
            float val = s[v] + Lin[(size_t)b * H_N + h];
            val = fmaxf(val, 0.0f) * uh;
            atomicAdd(out + b, val);
        }
    }
}

extern "C" void kernel_launch(void* const* d_in, const int* in_sizes, int n_in,
                              void* d_out, int out_size, void* d_ws, size_t ws_size,
                              hipStream_t stream) {
    const float* head = (const float*)d_in[0];
    // d_in[1] = relation_embeddings: unused by the reference computation
    const float* tail = (const float*)d_in[2];
    const float* W    = (const float*)d_in[3];
    const float* V    = (const float*)d_in[4];
    const float* u    = (const float*)d_in[5];
    const float* bR   = (const float*)d_in[6];
    float* out = (float*)d_out;

    char* ws = (char*)d_ws;
    unsigned short* headBF = (unsigned short*)ws;                          // 4 MiB
    float* Lin             = (float*)(ws + (size_t)B_N * D_N * 2);         // 4 MiB
    unsigned short* WtBF   = (unsigned short*)(ws + (size_t)B_N * D_N * 2
                                                  + (size_t)B_N * H_N * 4); // 128 MiB

    zero_out_kernel<<<(B_N + 255) / 256, 256, 0, stream>>>(out, B_N);
    convert_head_kernel<<<(B_N * D_N + 255) / 256, 256, 0, stream>>>(head, headBF, B_N * D_N);

    dim3 tgrid(D_N / 32, D_N / 32, H_N);
    dim3 tblk(32, 8);
    transpose_convert_W_kernel<<<tgrid, tblk, 0, stream>>>(W, WtBF);

    dim3 lgrid(B_N / 256, H_N);
    linear_kernel<<<lgrid, 256, 0, stream>>>(head, tail, V, bR, Lin);

    dim3 mgrid(H_N, B_N / 64);  // x: hidden unit, y: 64-batch group (4 waves x 16 rows)
    bilinear_wmma_kernel<<<mgrid, 128, 0, stream>>>(headBF, WtBF, tail, Lin, u, out);
}
